// BasicDeconvolutionBlock_13288628814172
// MI455X (gfx1250) — matrix operand order repaired and checked by reading, not compiled, not run on hardware
//
#include <hip/hip_runtime.h>
#include <hip/hip_bf16.h>

typedef float  v8f   __attribute__((ext_vector_type(8)));
typedef __bf16 v8bf  __attribute__((ext_vector_type(8)));
typedef __bf16 v16bf __attribute__((ext_vector_type(16)));

#define INC    256
#define OUTC   256
#define KTAPS  27
#define BM     64          // rows (points) per block tile
#define LDAB   264         // bf16 elements per LDS row (256 + 8 pad -> conflict-free A frags)
#define BN_EPS 1e-5f

static __device__ __forceinline__ unsigned short f2bf(float x) {
    unsigned u = __float_as_uint(x);
    u += 0x7fffu + ((u >> 16) & 1u);          // round-to-nearest-even
    return (unsigned short)(u >> 16);
}

// Repack W [27][INC][OUTC] f32 -> Wp [27][OUTC][INC] bf16 so that a B fragment
// (one output column) is two 8-value K runs, each contiguous in memory.
__global__ __launch_bounds__(256) void wpack_kernel(const float* __restrict__ W,
                                                    unsigned short* __restrict__ Wp) {
    int e  = blockIdx.x * 256 + threadIdx.x;   // e < 27*256*256
    int k  = e >> 16;
    int ci = (e >> 8) & 255;
    int co = e & 255;
    Wp[((k * OUTC + co) << 8) + ci] = f2bf(W[e]);
}

__global__ __launch_bounds__(256) void init_stats_kernel(float* __restrict__ ws) {
    ws[blockIdx.x * 256 + threadIdx.x] = 0.0f;  // ws[0..255]=sum, ws[256..511]=sumsq
}

// Fused: 27-tap gathered GEMM (bf16 WMMA, f32 accumulate) + per-channel sum/sumsq.
// Block: 256 threads = 8 wave32s, arranged 4 (M) x 2 (N). Block tile: 64 rows x 128 cols.
// Each wave: 16 rows x 64 cols = 4 accumulators of 16x16 f32.
__global__ __launch_bounds__(256) void spdeconv_gemm_kernel(
    const float* __restrict__ feats, const unsigned short* __restrict__ Wp,
    const int* __restrict__ nbr, float* __restrict__ y, float* __restrict__ ws, int N)
{
    __shared__ unsigned short As[BM * LDAB];   // 33 KB bf16 staging of gathered A tile
    __shared__ int   rowIdx[BM];
    __shared__ float red[2 * 128];

    const int tid   = threadIdx.x;
    const int lane  = tid & 31;
    const int wave  = tid >> 5;
    const int waveM = wave & 3;                // row sub-tile (16 rows each)
    const int waveN = wave >> 2;               // 0/1 -> 64-col half of the block tile
    const int lcol  = lane & 15;
    const int half  = lane >> 4;               // lane half-wave select
    const int rowBase = blockIdx.x * BM;
    const int colBlk  = blockIdx.y * 128;

    v8f acc[4] = {};

    for (int k = 0; k < KTAPS; ++k) {
        __syncthreads();                       // previous tap done with As
        if (tid < BM) {
            int r = rowBase + tid;
            rowIdx[tid] = (r < N) ? nbr[k * N + r] : -1;
        }
        __syncthreads();

        // Cooperative masked gather: 64 rows x 256 f32 -> bf16 in LDS.
        // Per step j, 64 consecutive threads cover one full 1KB row (coalesced b128 loads).
        #pragma unroll 4
        for (int j = 0; j < 16; ++j) {
            int f   = tid + (j << 8);
            int r   = f >> 6;
            int c4  = f & 63;
            int idx = rowIdx[r];
            float4 v = make_float4(0.f, 0.f, 0.f, 0.f);
            if (idx >= 0)
                v = *(const float4*)(feats + (size_t)idx * INC + (c4 << 2));
            unsigned long long pk =
                  (unsigned long long)f2bf(v.x)
                | ((unsigned long long)f2bf(v.y) << 16)
                | ((unsigned long long)f2bf(v.z) << 32)
                | ((unsigned long long)f2bf(v.w) << 48);
            *(unsigned long long*)&As[r * LDAB + (c4 << 2)] = pk;
        }
        __syncthreads();

        const unsigned short* Arow = &As[(waveM * 16 + lcol) * LDAB];
        #pragma unroll
        for (int s = 0; s < 8; ++s) {          // K steps of 32
            const int kb = s << 5;
            // A 16x32 bf16 fragment per ISA layout: lanes0-15 K[kb..kb+7]+K[kb+16..23],
            // lanes16-31 K[kb+8..15]+K[kb+24..31].
            v8bf alo = *reinterpret_cast<const v8bf*>(Arow + kb + half * 8);
            v8bf ahi = *reinterpret_cast<const v8bf*>(Arow + kb + 16 + half * 8);
            v16bf a = __builtin_shufflevector(alo, ahi,
                        0,1,2,3,4,5,6,7,8,9,10,11,12,13,14,15);
            #pragma unroll
            for (int t = 0; t < 4; ++t) {
                // B 32x16 fragment: col = lcol of this tile; lanes0-15 K[kb..kb+7]+K[kb+16..23],
                // lanes16-31 K[kb+8..15]+K[kb+24..31]; contiguous thanks to the Wp repack.
                const unsigned short* bp = Wp
                    + (((size_t)k * OUTC + colBlk + waveN * 64 + t * 16 + lcol) << 8)
                    + kb + half * 8;
                v8bf blo = *reinterpret_cast<const v8bf*>(bp);
                v8bf bhi = *reinterpret_cast<const v8bf*>(bp + 16);
                v16bf b = __builtin_shufflevector(blo, bhi,
                            0,1,2,3,4,5,6,7,8,9,10,11,12,13,14,15);
                acc[t] = __builtin_amdgcn_wmma_f32_16x16x32_bf16(
                    false, a, false, b, (short)0, acc[t], false, false);
            }
        }
    }

    // Epilogue: store pre-BN y and reduce per-channel sum / sumsq.
    red[tid] = 0.f;
    __syncthreads();
    #pragma unroll
    for (int t = 0; t < 4; ++t) {
        const float* ap = reinterpret_cast<const float*>(&acc[t]);
        int colLocal = waveN * 64 + t * 16 + lcol;       // 0..127
        float s = 0.f, q = 0.f;
        #pragma unroll
        for (int v = 0; v < 8; ++v) {
            int row = rowBase + waveM * 16 + half * 8 + v;   // C layout: M = v (+8 hi half)
            if (row < N) {
                float val = ap[v];
                y[(size_t)row * OUTC + colBlk + colLocal] = val;
                s += val;
                q += val * val;
            }
        }
        atomicAdd(&red[colLocal], s);
        atomicAdd(&red[128 + colLocal], q);
    }
    __syncthreads();
    // 256 entries: stat = tid>>7 (0=sum,1=sumsq), col = tid&127
    atomicAdd(&ws[((tid >> 7) << 8) + colBlk + (tid & 127)], red[tid]);
}

__global__ __launch_bounds__(256) void bn_finalize_kernel(float* __restrict__ ws,
                                                          const float* __restrict__ gamma,
                                                          const float* __restrict__ beta,
                                                          int N) {
    int c = threadIdx.x;
    float inv  = 1.0f / (float)N;
    float mean = ws[c] * inv;
    float var  = fmaxf(ws[256 + c] * inv - mean * mean, 0.f);
    float rstd = rsqrtf(var + BN_EPS);
    float scale = gamma[c] * rstd;
    ws[512 + c] = scale;
    ws[768 + c] = beta[c] - mean * scale;
}

__global__ __launch_bounds__(256) void bn_relu_kernel(float* __restrict__ y,
                                                      const float* __restrict__ ws, int n4) {
    int g = blockIdx.x * 256 + threadIdx.x;
    if (g >= n4) return;
    float4 v = ((const float4*)y)[g];
    int c = (g & 63) << 2;
    const float* sc = ws + 512;
    const float* sh = ws + 768;
    v.x = fmaxf(v.x * sc[c + 0] + sh[c + 0], 0.f);
    v.y = fmaxf(v.y * sc[c + 1] + sh[c + 1], 0.f);
    v.z = fmaxf(v.z * sc[c + 2] + sh[c + 2], 0.f);
    v.w = fmaxf(v.w * sc[c + 3] + sh[c + 3], 0.f);
    ((float4*)y)[g] = v;
}

extern "C" void kernel_launch(void* const* d_in, const int* in_sizes, int n_in,
                              void* d_out, int out_size, void* d_ws, size_t ws_size,
                              hipStream_t stream) {
    const float* feats = (const float*)d_in[0];
    const float* W     = (const float*)d_in[1];
    const float* gamma = (const float*)d_in[2];
    const float* beta  = (const float*)d_in[3];
    const int*   nbr   = (const int*)d_in[4];
    const int N = in_sizes[0] / INC;

    float* y  = (float*)d_out;
    float* ws = (float*)d_ws;                                   // [0..1023] f32 stats/params
    unsigned short* Wp = (unsigned short*)((char*)d_ws + 4096); // 27*256*256 bf16 (~3.5 MB)

    wpack_kernel<<<(KTAPS * INC * OUTC) / 256, 256, 0, stream>>>(W, Wp);
    init_stats_kernel<<<2, 256, 0, stream>>>(ws);

    dim3 grid((N + BM - 1) / BM, 2);
    spdeconv_gemm_kernel<<<grid, 256, 0, stream>>>(feats, Wp, nbr, y, ws, N);

    bn_finalize_kernel<<<1, 256, 0, stream>>>(ws, gamma, beta, N);
    bn_relu_kernel<<<(N * 64 + 255) / 256, 256, 0, stream>>>(y, ws, N * 64);
}
